// Model_30219389895259
// MI455X (gfx1250) — matrix operand order repaired and checked by reading, not compiled, not run on hardware
//
#include <hip/hip_runtime.h>
#include <cstdint>
#include <cstddef>

// ---------------------------------------------------------------------------
// Model dims (fixed by the reference)
// ---------------------------------------------------------------------------
constexpr int B_  = 4;
constexpr int N_  = 4096;
constexpr int KNB = 16;   // neighbors
constexpr int CIN = 16;
constexpr int C64 = 64;   // D == 2*COUT == 64

typedef _Float16 f16;
typedef __attribute__((ext_vector_type(16))) _Float16 v16h;
typedef __attribute__((ext_vector_type(8)))  _Float16 v8h;
typedef __attribute__((ext_vector_type(8)))  float    v8f;

union HV { v16h v; v8h h[2]; f16 e[16]; };

#define DEVFN static __device__ __forceinline__

// D = A(16x32 f16) x B(32x16 f16) + C(16x16 f32)   -- wave32 WMMA
DEVFN v8f wmma16(v16h a, v16h b, v8f c) {
  return __builtin_amdgcn_wmma_f32_16x16x32_f16(false, a, false, b, (short)0, c, false, false);
}

// wave-local LDS fence: DS ops issue in order per wave; waiting DScnt==0 makes
// this wave's staged LDS data visible to all its lanes (no block barrier).
DEVFN void wave_lds_fence() { asm volatile("s_wait_dscnt 0x0" ::: "memory"); }

DEVFN float xor_max16(float v) {
  #pragma unroll
  for (int o = 8; o > 0; o >>= 1) v = fmaxf(v, __shfl_xor(v, o, 16));
  return v;
}
DEVFN float xor_add16(float v) {
  #pragma unroll
  for (int o = 8; o > 0; o >>= 1) v += __shfl_xor(v, o, 16);
  return v;
}
// merge softmax partial accumulators (m = running max, s = sum of exp(x-m))
// across the 16-lane group; executed once per row, outside the stream loop.
DEVFN void merge16(float& m, float& s) {
  #pragma unroll
  for (int o = 8; o > 0; o >>= 1) {
    float om = __shfl_xor(m, o, 16);
    float os = __shfl_xor(s, o, 16);
    float mn = fmaxf(m, om);
    s = s * __expf(m - mn) + os * __expf(om - mn);
    m = mn;
  }
}

// ---- fragment loaders (contiguous K order k = 16*hi + s, same in A and B) ----------------------
// A fragment: rows (lane&15)+row0 ; per-lane 16 contiguous K {0..15} (lane<16)
// or {16..31} (lane>=16) of the 32-wide K chunk. Row-major f16, `stride`.
DEVFN v16h fragA_wh(const f16* Wh, int stride, int row0, int ks, int lane) {
  int hi = lane >> 4;
  const f16* p = Wh + (size_t)(row0 + (lane & 15)) * stride + ks * 32 + hi * 16;
  HV u; u.h[0] = *(const v8h*)p; u.h[1] = *(const v8h*)(p + 8); return u.v;
}
// A fragment where K runs along a dim with row stride N_ (used for V tiles).
DEVFN v16h fragA_strideN(const f16* base, int row0, int kbase, int lane) {
  int hi = lane >> 4;
  const f16* p = base + (size_t)(row0 + (lane & 15)) * N_ + kbase + hi * 16;
  HV u; u.h[0] = *(const v8h*)p; u.h[1] = *(const v8h*)(p + 8); return u.v;
}
// B fragment: col (lane&15)+col0 ; per-lane 16 contiguous K (hi selects 16..31)
// from column-major-per-point storage [col][64].
DEVFN v16h fragB_cm64(const f16* base, int col0, int ks, int lane) {
  int hi = lane >> 4;
  const f16* p = base + (size_t)(col0 + (lane & 15)) * 64 + ks * 32 + hi * 16;
  HV u; u.h[0] = *(const v8h*)p; u.h[1] = *(const v8h*)(p + 8); return u.v;
}
// B fragment from an LDS staging buffer laid out [col][stride].
DEVFN v16h fragB_lds(const f16* st, int stride, int ks, int lane) {
  int hi = lane >> 4;
  const f16* p = st + (lane & 15) * stride + ks * 32 + hi * 16;
  HV u; u.h[0] = *(const v8h*)p; u.h[1] = *(const v8h*)(p + 8); return u.v;
}

// ---------------------------------------------------------------------------
// 1) fea_fuse: gather neighbors, two tiny MLP+BN+ReLU, max over K -> fea
// ---------------------------------------------------------------------------
__global__ __launch_bounds__(64)
void k_feafuse(const float* __restrict__ feat, const float* __restrict__ xyz,
               const int* __restrict__ idx,
               const float* __restrict__ w1, const float* __restrict__ w2,
               const float* __restrict__ s1, const float* __restrict__ b1,
               const float* __restrict__ s2, const float* __restrict__ b2,
               float* __restrict__ fea) {
  __shared__ float scf[CIN];
  __shared__ float sgf[KNB * CIN];
  __shared__ float scx[3];
  __shared__ float sgx[KNB * 3];
  __shared__ int   sidx[KNB];
  int t = threadIdx.x, n = blockIdx.x, b = blockIdx.y;
  if (t < KNB) sidx[t] = idx[((size_t)b * KNB + t) * N_ + n];
  __syncthreads();
  if (t < CIN) scf[t] = feat[((size_t)b * CIN + t) * N_ + n];
  if (t < 3)   scx[t] = xyz[((size_t)b * 3 + t) * N_ + n];
  for (int e = t; e < KNB * CIN; e += 64) {
    int kk = e >> 4, c = e & 15;
    sgf[e] = feat[((size_t)b * CIN + c) * N_ + sidx[kk]];
  }
  for (int e = t; e < KNB * 3; e += 64) {
    int kk = e / 3, c = e % 3;
    sgx[e] = xyz[((size_t)b * 3 + c) * N_ + sidx[kk]];
  }
  __syncthreads();
  int o = t;
  float best = -3.0e38f;
  if (o < 32) {
    float w[6];
    #pragma unroll
    for (int c = 0; c < 6; c++) w[c] = w1[o * 6 + c];
    float sc = s1[o], sb = b1[o];
    for (int kk = 0; kk < KNB; kk++) {
      float val = w[0]*scx[0] + w[1]*scx[1] + w[2]*scx[2]
                + w[3]*(sgx[kk*3+0]-scx[0]) + w[4]*(sgx[kk*3+1]-scx[1]) + w[5]*(sgx[kk*3+2]-scx[2]);
      best = fmaxf(best, fmaxf(val * sc + sb, 0.f));
    }
  } else {
    int oo = o - 32;
    float wA[CIN], wB[CIN];
    #pragma unroll
    for (int c = 0; c < CIN; c++) { wA[c] = w2[oo*32 + c]; wB[c] = w2[oo*32 + 16 + c]; }
    float sc = s2[oo], sb = b2[oo];
    float basev = 0.f;
    #pragma unroll
    for (int c = 0; c < CIN; c++) basev += wA[c] * scf[c];
    for (int kk = 0; kk < KNB; kk++) {
      float val = basev;
      #pragma unroll
      for (int c = 0; c < CIN; c++) val += wB[c] * (sgf[kk*16 + c] - scf[c]);
      best = fmaxf(best, fmaxf(val * sc + sb, 0.f));
    }
  }
  fea[((size_t)b * C64 + o) * N_ + n] = best;
}

// ---------------------------------------------------------------------------
// 2) Generic 64x64 WMMA GEMM over (B,64,N), fusion flags as template consts:
//    Y = [res +] act( W @ (Xa [+|-] Xb) + bias [*bn_s + bn_b] )
// ---------------------------------------------------------------------------
enum { F_SUB = 1, F_ADD = 2, F_BIAS = 4, F_BN = 8, F_RELU = 16, F_RES = 32 };

template <int FLAGS>
__global__ __launch_bounds__(256)
void k_gemm64(const float* __restrict__ W, const float* __restrict__ bias,
              const float* __restrict__ bns, const float* __restrict__ bnb,
              const float* __restrict__ Xa, const float* __restrict__ Xb,
              const float* __restrict__ R, float* __restrict__ Y) {
  int tid = threadIdx.x, wv = tid >> 5, lane = tid & 31, hi = lane >> 4;
  int b = blockIdx.y;
  int rt = wv & 3, ct = wv >> 2;
  int col = blockIdx.x * 32 + ct * 16 + (lane & 15);
  size_t base = (size_t)b * C64 * N_;
  // A fragments from f32 weights (64x64 row-major), converted to f16.
  v16h Aw[2];
  #pragma unroll
  for (int ks = 0; ks < 2; ks++) {
    const float* p = W + (size_t)(rt * 16 + (lane & 15)) * 64 + ks * 32 + hi * 8;
    HV u;
    #pragma unroll
    for (int e = 0; e < 8; e++) { u.e[e] = (f16)p[e]; u.e[8 + e] = (f16)p[16 + e]; }
    Aw[ks] = u.v;
  }
  v8f acc = {};
  #pragma unroll
  for (int ks = 0; ks < 2; ks++) {
    float xa[16], xb[16];
    #pragma unroll
    for (int e = 0; e < 16; e++) {
      int c = ks * 32 + hi * 8 + (e & 7) + ((e >> 3) << 4);
      xa[e] = Xa[base + (size_t)c * N_ + col];
      if (FLAGS & (F_SUB | F_ADD)) xb[e] = Xb[base + (size_t)c * N_ + col];
    }
    HV u;
    #pragma unroll
    for (int e = 0; e < 16; e++) {
      float xv = xa[e];
      if (FLAGS & F_SUB) xv -= xb[e];
      if (FLAGS & F_ADD) xv += xb[e];
      u.e[e] = (f16)xv;
    }
    acc = wmma16(Aw[ks], u.v, acc);
  }
  #pragma unroll
  for (int r = 0; r < 8; r++) {
    int row = rt * 16 + hi * 8 + r;
    float yv = acc[r];
    if (FLAGS & F_BIAS) yv += bias[row];
    if (FLAGS & F_BN)   yv = yv * bns[row] + bnb[row];
    if (FLAGS & F_RELU) yv = fmaxf(yv, 0.f);
    if (FLAGS & F_RES)  yv += R[base + (size_t)row * N_ + col];
    Y[base + (size_t)row * N_ + col] = yv;
  }
}

// ---------------------------------------------------------------------------
// 3) pack q,k -> per-point-column f16 (B,N,64); v -> f16 (B,64,N)
// ---------------------------------------------------------------------------
__global__ __launch_bounds__(256)
void k_pack(const float* __restrict__ q, const float* __restrict__ k,
            const float* __restrict__ v,
            f16* __restrict__ qt, f16* __restrict__ kt, f16* __restrict__ vh) {
  size_t id = (size_t)blockIdx.x * blockDim.x + threadIdx.x;  // B*64*N threads
  size_t b = id / ((size_t)C64 * N_);
  size_t rem = id % ((size_t)C64 * N_);
  size_t d = rem / N_, n = rem % N_;
  size_t tp = (b * N_ + n) * 64 + d;
  qt[tp] = (f16)q[id];
  kt[tp] = (f16)k[id];
  vh[id] = (f16)v[id];
}

// pack local-branch weights to padded f16 tiles
__global__ __launch_bounds__(256)
void k_packw(const float* __restrict__ fd_w1, const float* __restrict__ fd_w2,
             const float* __restrict__ fg_w1, const float* __restrict__ fg_w2,
             f16* __restrict__ wd1p, f16* __restrict__ wd2p,
             f16* __restrict__ wg1p, f16* __restrict__ wg2p) {
  int id = blockIdx.x * blockDim.x + threadIdx.x;
  if (id < 64 * 32) { int o = id >> 5, c = id & 31; wd1p[id] = (f16)(c < 10 ? fd_w1[o * 10 + c] : 0.f); }
  if (id < 64 * 64)  wd2p[id] = (f16)fd_w2[id];
  if (id < 256 * 64) wg1p[id] = (f16)fg_w1[id];
  if (id < 64 * 256) wg2p[id] = (f16)fg_w2[id];
}

// ---------------------------------------------------------------------------
// 4) global-attention pass A: row softmax stats via WMMA streaming.
//    Lane-private online (m,s) accumulators -> NO cross-lane ops in the loop;
//    one (m,s)-merge across the 16-lane group at the end.
//    Stores fused logit offset L = m + log(s)  (so pass B does exp(E - L)).
// ---------------------------------------------------------------------------
__global__ __launch_bounds__(256)
void k_rowstats(const f16* __restrict__ qt, const f16* __restrict__ kt,
                float* __restrict__ lrow) {
  int tid = threadIdx.x, wv = tid >> 5, lane = tid & 31, hi = lane >> 4;
  int b = blockIdx.y;
  int j0 = (blockIdx.x * 8 + wv) * 16;
  const f16* qb = qt + (size_t)b * N_ * 64;
  const f16* kb = kt + (size_t)b * N_ * 64;
  v16h A0 = fragA_wh(qb, 64, j0, 0, lane);
  v16h A1 = fragA_wh(qb, 64, j0, 1, lane);
  float pm[8], ps[8];
  #pragma unroll
  for (int r = 0; r < 8; r++) { pm[r] = -3.0e38f; ps[r] = 0.f; }
  for (int it = 0; it < N_ / 16; it++) {
    if (it + 8 < N_ / 16)
      __builtin_prefetch((const void*)(kb + (size_t)(it + 8) * 16 * 64 + (lane << 5)), 0, 1);
    v16h B0 = fragB_cm64(kb, it * 16, 0, lane);
    v16h B1 = fragB_cm64(kb, it * 16, 1, lane);
    v8f E = {};
    E = wmma16(A0, B0, E);
    E = wmma16(A1, B1, E);
    #pragma unroll
    for (int r = 0; r < 8; r++) {
      float val = E[r];
      float mn = fmaxf(pm[r], val);
      ps[r] = ps[r] * __expf(pm[r] - mn) + __expf(val - mn);
      pm[r] = mn;
    }
  }
  #pragma unroll
  for (int r = 0; r < 8; r++) merge16(pm[r], ps[r]);
  if ((lane & 15) == 0) {
    #pragma unroll
    for (int r = 0; r < 8; r++) {
      int jj = j0 + hi * 8 + r;
      lrow[(size_t)b * N_ + jj] = pm[r] + __logf(ps[r]);
    }
  }
}

// ---------------------------------------------------------------------------
// 5) pass B: x_r[:,i] = (1/(1e-9+c_i)) * sum_j exp(q_j.k_i - L_j) * v[:,j]
//    One wave owns a 16-col tile, streams j in 32-chunks; E tiles via WMMA,
//    repacked through per-wave LDS into B-fragments for the V x W WMMA.
// ---------------------------------------------------------------------------
__global__ __launch_bounds__(256)
void k_attnout(const f16* __restrict__ qt, const f16* __restrict__ kt,
               const f16* __restrict__ vh, const float* __restrict__ lrow,
               float* __restrict__ xr) {
  __shared__ f16 stg[8 * 512];
  int tid = threadIdx.x, wv = tid >> 5, lane = tid & 31, hi = lane >> 4;
  int b = blockIdx.y;
  int i0 = (blockIdx.x * 8 + wv) * 16;
  f16* st = stg + wv * 512;
  const f16* qb = qt + (size_t)b * N_ * 64;
  const f16* kb = kt + (size_t)b * N_ * 64;
  const f16* vb = vh + (size_t)b * C64 * N_;
  const float* lb = lrow + (size_t)b * N_;
  v16h Bk0 = fragB_cm64(kb, i0, 0, lane);
  v16h Bk1 = fragB_cm64(kb, i0, 1, lane);
  v8f acc[4] = {};
  float cs = 0.f;
  for (int jc = 0; jc < N_ / 32; jc++) {
    int jb0 = jc * 32;
    if (jc + 4 < N_ / 32)
      __builtin_prefetch((const void*)(qb + (size_t)(jb0 + 128) * 64 + (lane << 6)), 0, 1);
    v8f E0 = {}, E1 = {};
    E0 = wmma16(fragA_wh(qb, 64, jb0,      0, lane), Bk0, E0);
    E0 = wmma16(fragA_wh(qb, 64, jb0,      1, lane), Bk1, E0);
    E1 = wmma16(fragA_wh(qb, 64, jb0 + 16, 0, lane), Bk0, E1);
    E1 = wmma16(fragA_wh(qb, 64, jb0 + 16, 1, lane), Bk1, E1);
    #pragma unroll
    for (int r = 0; r < 8; r++) {
      int jr = jb0 + hi * 8 + r;
      float w0 = __expf(E0[r] - lb[jr]);
      cs += w0;
      st[(lane & 15) * 32 + hi * 8 + r] = (f16)w0;
      float w1 = __expf(E1[r] - lb[jr + 16]);
      cs += w1;
      st[(lane & 15) * 32 + 16 + hi * 8 + r] = (f16)w1;
    }
    wave_lds_fence();               // per-wave staging: DScnt==0 is sufficient
    v16h Bw = fragB_lds(st, 32, 0, lane);
    wave_lds_fence();               // reads done before next iter overwrites
    #pragma unroll
    for (int rt = 0; rt < 4; rt++)
      acc[rt] = wmma16(fragA_strideN(vb, rt * 16, jb0, lane), Bw, acc[rt]);
  }
  float ct = cs + __shfl_xor(cs, 16, 32);
  float inv = 1.f / (1e-9f + ct);
  #pragma unroll
  for (int rt = 0; rt < 4; rt++)
    #pragma unroll
    for (int r = 0; r < 8; r++) {
      int d = rt * 16 + hi * 8 + r;
      xr[(size_t)b * C64 * N_ + (size_t)d * N_ + i0 + (lane & 15)] = acc[rt][r] * inv;
    }
}

// ---------------------------------------------------------------------------
// 6) fused local branch: per point (one wave): pos-enc MLP (10->64->64),
//    gate MLP (64->256->64), softmax over K=16 neighbors, weighted sum -> res
// ---------------------------------------------------------------------------
__global__ __launch_bounds__(128)
void k_local(const float* __restrict__ q, const float* __restrict__ k,
             const float* __restrict__ v, const float* __restrict__ xyz,
             const int* __restrict__ idx,
             const f16* __restrict__ wd1p, const f16* __restrict__ wd2p,
             const f16* __restrict__ wg1p, const f16* __restrict__ wg2p,
             const float* __restrict__ fd_b1, const float* __restrict__ fd_b2,
             const float* __restrict__ fg_b1, const float* __restrict__ fg_b2,
             float* __restrict__ res) {
  __shared__ f16 stg[4 * 4096];
  int tid = threadIdx.x, wv = tid >> 5, lane = tid & 31;
  int colk = lane & 15, hi = lane >> 4;
  int b = blockIdx.y, n = blockIdx.x * 4 + wv;
  f16* st = stg + wv * 4096;          // strictly per-wave staging
  size_t xb = (size_t)b * 3 * N_;
  size_t qb = (size_t)b * C64 * N_;
  int j = idx[((size_t)b * KNB + colk) * N_ + n];
  float cx = xyz[xb + 0 * N_ + n], cy = xyz[xb + 1 * N_ + n], cz = xyz[xb + 2 * N_ + n];
  float gx = xyz[xb + 0 * N_ + j], gy = xyz[xb + 1 * N_ + j], gz = xyz[xb + 2 * N_ + j];
  float rx = gx - cx, ry = gy - cy, rz = gz - cz;
  float nr = sqrtf(rx * rx + ry * ry + rz * rz + 1e-12f);
  // rel_pos B-fragment: channels 0..9 live, rest zero-padded to K=32.
  HV rb;
  #pragma unroll
  for (int e = 0; e < 16; e++) rb.e[e] = (f16)0.f;
  if (hi == 0) {
    rb.e[0] = (f16)cx; rb.e[1] = (f16)cy; rb.e[2] = (f16)cz;
    rb.e[3] = (f16)gx; rb.e[4] = (f16)gy; rb.e[5] = (f16)gz;
    rb.e[6] = (f16)rx; rb.e[7] = (f16)ry; rb.e[8] = (f16)rz;
    rb.e[9] = (f16)nr;
  }
  v16h Brel = rb.v;
  // h1 = relu(fd_w1 @ rel_pos + fd_b1)
  v8f hc[4];
  #pragma unroll
  for (int rt = 0; rt < 4; rt++) {
    v8f z = {};
    hc[rt] = wmma16(fragA_wh(wd1p, 32, rt * 16, 0, lane), Brel, z);
    #pragma unroll
    for (int r = 0; r < 8; r++) {
      int row = rt * 16 + hi * 8 + r;
      hc[rt][r] = fmaxf(hc[rt][r] + fd_b1[row], 0.f);
    }
  }
  #pragma unroll
  for (int rt = 0; rt < 4; rt++)
    #pragma unroll
    for (int r = 0; r < 8; r++)
      st[colk * 64 + rt * 16 + hi * 8 + r] = (f16)hc[rt][r];
  wave_lds_fence();
  // pos = fd_w2 @ h1 + fd_b2
  v8f pc[4];
  #pragma unroll
  for (int rt = 0; rt < 4; rt++) {
    v8f a = {};
    #pragma unroll
    for (int ks = 0; ks < 2; ks++)
      a = wmma16(fragA_wh(wd2p, 64, rt * 16, ks, lane), fragB_lds(st, 64, ks, lane), a);
    #pragma unroll
    for (int r = 0; r < 8; r++) pc[rt][r] = a[r] + fd_b2[rt * 16 + hi * 8 + r];
  }
  wave_lds_fence();
  // t = q - k_loc + pos (staged), vp = v_loc + pos (kept in regs)
  v8f vp[4];
  #pragma unroll
  for (int rt = 0; rt < 4; rt++)
    #pragma unroll
    for (int r = 0; r < 8; r++) {
      int row = rt * 16 + hi * 8 + r;
      float qv = q[qb + (size_t)row * N_ + n];
      float kv = k[qb + (size_t)row * N_ + j];
      float vv = v[qb + (size_t)row * N_ + j];
      vp[rt][r] = vv + pc[rt][r];
      st[colk * 64 + row] = (f16)(qv - kv + pc[rt][r]);
    }
  wave_lds_fence();
  // a1 = relu(fg_w1 @ t + fg_b1)   (256 rows)
  v16h Bt0 = fragB_lds(st, 64, 0, lane);
  v16h Bt1 = fragB_lds(st, 64, 1, lane);
  v8f a1[16];
  #pragma unroll
  for (int rt = 0; rt < 16; rt++) {
    v8f a = {};
    a = wmma16(fragA_wh(wg1p, 64, rt * 16, 0, lane), Bt0, a);
    a = wmma16(fragA_wh(wg1p, 64, rt * 16, 1, lane), Bt1, a);
    #pragma unroll
    for (int r = 0; r < 8; r++)
      a[r] = fmaxf(a[r] + fg_b1[rt * 16 + hi * 8 + r], 0.f);
    a1[rt] = a;
  }
  wave_lds_fence();
  #pragma unroll
  for (int rt = 0; rt < 16; rt++)
    #pragma unroll
    for (int r = 0; r < 8; r++)
      st[colk * 256 + rt * 16 + hi * 8 + r] = (f16)a1[rt][r];
  wave_lds_fence();
  // a = fg_w2 @ a1 + fg_b2
  v8f ac[4];
  #pragma unroll
  for (int rt = 0; rt < 4; rt++) {
    v8f a = {};
    #pragma unroll
    for (int ks = 0; ks < 8; ks++)
      a = wmma16(fragA_wh(wg2p, 256, rt * 16, ks, lane), fragB_lds(st, 256, ks, lane), a);
    #pragma unroll
    for (int r = 0; r < 8; r++) a[r] += fg_b2[rt * 16 + hi * 8 + r];
    ac[rt] = a;
  }
  // softmax over the 16 neighbor columns (lanes within each 16-group),
  // then res[d] = sum_k a~[d,k] * vp[d,k]
  #pragma unroll
  for (int rt = 0; rt < 4; rt++)
    #pragma unroll
    for (int r = 0; r < 8; r++) {
      float val = ac[rt][r];
      float mx = xor_max16(val);
      float ex = __expf(val - mx);
      float sm = xor_add16(ex);
      float w  = ex / sm;
      float red = xor_add16(w * vp[rt][r]);
      if (colk == 0) {
        int row = rt * 16 + hi * 8 + r;
        res[qb + (size_t)row * N_ + n] = red;
      }
    }
}

// ---------------------------------------------------------------------------
// launch
// ---------------------------------------------------------------------------
extern "C" void kernel_launch(void* const* d_in, const int* in_sizes, int n_in,
                              void* d_out, int out_size, void* d_ws, size_t ws_size,
                              hipStream_t stream) {
  (void)in_sizes; (void)n_in; (void)out_size; (void)ws_size;
  const float* feature = (const float*)d_in[0];
  const float* xyz     = (const float*)d_in[1];
  const int*   idx     = (const int*)  d_in[2];
  const float* ff_w1 = (const float*)d_in[3];
  const float* ff_w2 = (const float*)d_in[4];
  const float* c1_w  = (const float*)d_in[5];
  const float* wq_w  = (const float*)d_in[6];
  const float* wk_w  = (const float*)d_in[7];
  const float* wv_w  = (const float*)d_in[8];
  const float* fd_w1 = (const float*)d_in[9];
  const float* fd_w2 = (const float*)d_in[10];
  const float* fg_w1 = (const float*)d_in[11];
  const float* fg_w2 = (const float*)d_in[12];
  const float* tc_w  = (const float*)d_in[13];
  const float* c2_w  = (const float*)d_in[14];
  const float* c3_w  = (const float*)d_in[15];
  const float* lbr_w = (const float*)d_in[16];
  const float* c1_b  = (const float*)d_in[17];
  const float* wq_b  = (const float*)d_in[18];
  const float* wk_b  = (const float*)d_in[19];
  const float* wv_b  = (const float*)d_in[20];
  const float* fd_b1 = (const float*)d_in[21];
  const float* fd_b2 = (const float*)d_in[22];
  const float* fg_b1 = (const float*)d_in[23];
  const float* fg_b2 = (const float*)d_in[24];
  const float* tc_b  = (const float*)d_in[25];
  const float* c2_b  = (const float*)d_in[26];
  const float* c3_b  = (const float*)d_in[27];
  const float* lbr_b = (const float*)d_in[28];
  const float* bn1s  = (const float*)d_in[29];
  const float* bn1b  = (const float*)d_in[30];
  const float* bn2s  = (const float*)d_in[31];
  const float* bn2b  = (const float*)d_in[32];
  const float* an_s  = (const float*)d_in[33];
  const float* an_b  = (const float*)d_in[34];
  const float* lbrs  = (const float*)d_in[35];
  const float* lbrb  = (const float*)d_in[36];

  const size_t SZ = (size_t)B_ * C64 * N_;  // 1,048,576 elements
  uint8_t* w = (uint8_t*)d_ws;
  auto allocF = [&](size_t n) { float* p = (float*)w; w += n * sizeof(float); return p; };
  auto allocH = [&](size_t n) { f16*   p = (f16*)w;   w += n * sizeof(f16);   return p; };

  float* fea  = allocF(SZ);
  float* x    = allocF(SZ);
  float* q    = allocF(SZ);
  float* k    = allocF(SZ);
  float* v    = allocF(SZ);
  float* xr   = allocF(SZ);
  float* u    = allocF(SZ);
  float* resg = allocF(SZ);
  float* res  = allocF(SZ);
  float* resl = allocF(SZ);
  float* lrow = allocF((size_t)B_ * N_);
  f16* qt   = allocH(SZ);
  f16* kt   = allocH(SZ);
  f16* vh   = allocH(SZ);
  f16* wd1p = allocH(64 * 32);
  f16* wd2p = allocH(64 * 64);
  f16* wg1p = allocH(256 * 64);
  f16* wg2p = allocH(64 * 256);

  dim3 gN32(N_ / 32, B_);

  k_packw<<<64, 256, 0, stream>>>(fd_w1, fd_w2, fg_w1, fg_w2, wd1p, wd2p, wg1p, wg2p);
  k_feafuse<<<dim3(N_, B_), 64, 0, stream>>>(feature, xyz, idx, ff_w1, ff_w2,
                                             bn1s, bn1b, bn2s, bn2b, fea);
  // x = c1(fea); q,k,v from x
  k_gemm64<F_BIAS><<<gN32, 256, 0, stream>>>(c1_w, c1_b, nullptr, nullptr, fea, nullptr, nullptr, x);
  k_gemm64<F_BIAS><<<gN32, 256, 0, stream>>>(wq_w, wq_b, nullptr, nullptr, x, nullptr, nullptr, q);
  k_gemm64<F_BIAS><<<gN32, 256, 0, stream>>>(wk_w, wk_b, nullptr, nullptr, x, nullptr, nullptr, k);
  k_gemm64<F_BIAS><<<gN32, 256, 0, stream>>>(wv_w, wv_b, nullptr, nullptr, x, nullptr, nullptr, v);
  k_pack<<<(unsigned)(SZ / 256), 256, 0, stream>>>(q, k, v, qt, kt, vh);
  // global attention (two streaming passes, no NxN materialization)
  k_rowstats<<<dim3(N_ / 16 / 8, B_), 256, 0, stream>>>(qt, kt, lrow);
  k_attnout<<<dim3(N_ / 16 / 8, B_), 256, 0, stream>>>(qt, kt, vh, lrow, xr);
  // u = relu(an(tc(x - x_r))) ; res_global = c3(x + u)
  k_gemm64<F_SUB | F_BIAS | F_BN | F_RELU><<<gN32, 256, 0, stream>>>(
      tc_w, tc_b, an_s, an_b, x, xr, nullptr, u);
  k_gemm64<F_ADD | F_BIAS><<<gN32, 256, 0, stream>>>(
      c3_w, c3_b, nullptr, nullptr, x, u, nullptr, resg);
  // local branch -> res ; res_local = c2(res)
  k_local<<<dim3(N_ / 4, B_), 128, 0, stream>>>(q, k, v, xyz, idx, wd1p, wd2p, wg1p, wg2p,
                                                fd_b1, fd_b2, fg_b1, fg_b2, res);
  k_gemm64<F_BIAS><<<gN32, 256, 0, stream>>>(c2_w, c2_b, nullptr, nullptr, res, nullptr, nullptr, resl);
  // out = fea + relu(bn(lbr(res_local + res_global)))
  k_gemm64<F_ADD | F_BIAS | F_BN | F_RELU | F_RES><<<gN32, 256, 0, stream>>>(
      lbr_w, lbr_b, lbrs, lbrb, resl, resg, fea, (float*)d_out);
}
